// EEGGAT_55937654063613
// MI455X (gfx1250) — matrix operand order repaired and checked by reading, not compiled, not run on hardware
//
#include <hip/hip_runtime.h>
#include <hip/hip_bf16.h>
#include <math.h>

#define N_NODES    50000
#define N_EDGES    800000
#define IN_CH      128
#define HID        64
#define NUM_GRAPHS 512
#define NEG_SLOPE  0.2f
#define E_TOT      (N_EDGES + N_NODES)   // edges + self loops

typedef __attribute__((ext_vector_type(16))) _Float16 v16h;
typedef __attribute__((ext_vector_type(8)))  float    v8f;

// ---------------------------------------------------------------- utilities

__global__ void k_fill(float* __restrict__ p, float v, int n) {
    int i = blockIdx.x * blockDim.x + threadIdx.x;
    if (i < n) p[i] = v;
}

__device__ __forceinline__ float atomicMaxFloat(float* addr, float value) {
    // ordering trick: valid for mixed signs given init = -inf (0xFF800000)
    if (value >= 0.0f)
        return __int_as_float(atomicMax((int*)addr, __float_as_int(value)));
    else
        return __uint_as_float(atomicMin((unsigned int*)addr, __float_as_uint(value)));
}

// ---------------------------------------------------------------- W pre-pack
// Repack W[K,64] (f32, row-major) into f16 WMMA B-fragment order:
// Bp[((kt*4 + g)*32 + lane)*16 + e] = W[(kt*32 + (lane>=16?8:0) + e + (e>=8?8:0))*64 + g*16 + (lane&15)]
__global__ void k_pack_w(const float* __restrict__ W, _Float16* __restrict__ Bp, int K) {
    int t = blockIdx.x * blockDim.x + threadIdx.x;
    if (t >= K * HID) return;
    int e    = t & 15;
    int lane = (t >> 4) & 31;
    int g    = (t >> 9) & 3;
    int kt   = t >> 11;
    int kk   = kt * 32 + ((lane >= 16) ? 8 : 0) + e + ((e >= 8) ? 8 : 0);
    int col  = g * 16 + (lane & 15);
    Bp[t] = (_Float16)W[kk * HID + col];
}

// ---------------------------------------------------------------- WMMA GEMM
// H[M,64] = X[M,K] @ W[K,64]  (f16 in, f32 accumulate)
// One wave = one 16-row M tile x 64 columns (4 accumulators).
// B comes pre-packed in fragment order (v16h per lane per frag).
__global__ void k_gemm_wmma(const float* __restrict__ X,
                            const _Float16* __restrict__ Bpack,
                            float* __restrict__ H, int M, int K) {
    const int lane = threadIdx.x & 31;
    const int wave = threadIdx.x >> 5;
    int mtile = blockIdx.x * (blockDim.x >> 5) + wave;
    const int ntiles = M >> 4;               // M multiple of 16
    if (mtile >= ntiles) mtile = ntiles - 1; // clamp: keep EXEC all-ones for WMMA

    const int mrow  = (mtile << 4) + (lane & 15);
    const int koffA = (lane >= 16) ? 8 : 0;  // A frag K sub-offset per half-wave
    const int ncol  = lane & 15;
    const v16h* bp  = (const v16h*)Bpack;

    v8f acc0 = {}, acc1 = {}, acc2 = {}, acc3 = {};

    for (int kt = 0, ktb = 0; kt < K; kt += 32, ktb += 4) {
        // A fragment: two contiguous 8-float runs -> 4x b128 loads
        const float* xr = X + mrow * K + kt + koffA;
        float4 r0 = *(const float4*)(xr + 0);
        float4 r1 = *(const float4*)(xr + 4);
        float4 q0 = *(const float4*)(xr + 16);
        float4 q1 = *(const float4*)(xr + 20);
        v16h a;
        a[0]  = (_Float16)r0.x; a[1]  = (_Float16)r0.y; a[2]  = (_Float16)r0.z; a[3]  = (_Float16)r0.w;
        a[4]  = (_Float16)r1.x; a[5]  = (_Float16)r1.y; a[6]  = (_Float16)r1.z; a[7]  = (_Float16)r1.w;
        a[8]  = (_Float16)q0.x; a[9]  = (_Float16)q0.y; a[10] = (_Float16)q0.z; a[11] = (_Float16)q0.w;
        a[12] = (_Float16)q1.x; a[13] = (_Float16)q1.y; a[14] = (_Float16)q1.z; a[15] = (_Float16)q1.w;

        // B fragments: one aligned 32B load each (L0-resident, shared by all waves)
        v16h b0 = bp[(ktb + 0) * 32 + lane];
        v16h b1 = bp[(ktb + 1) * 32 + lane];
        v16h b2 = bp[(ktb + 2) * 32 + lane];
        v16h b3 = bp[(ktb + 3) * 32 + lane];

        acc0 = __builtin_amdgcn_wmma_f32_16x16x32_f16(false, a, false, b0, (short)0, acc0, false, false);
        acc1 = __builtin_amdgcn_wmma_f32_16x16x32_f16(false, a, false, b1, (short)0, acc1, false, false);
        acc2 = __builtin_amdgcn_wmma_f32_16x16x32_f16(false, a, false, b2, (short)0, acc2, false, false);
        acc3 = __builtin_amdgcn_wmma_f32_16x16x32_f16(false, a, false, b3, (short)0, acc3, false, false);
    }

    // D layout: VGPR r holds row M = mtile*16 + r + (lane>=16 ? 8 : 0), col = lane&15
    const int mbase = (mtile << 4) + ((lane >= 16) ? 8 : 0);
#pragma unroll
    for (int r = 0; r < 8; ++r) {
        float* hr = H + (mbase + r) * HID + ncol;
        hr[0]  = acc0[r];
        hr[16] = acc1[r];
        hr[32] = acc2[r];
        hr[48] = acc3[r];
    }
}

// ---------------------------------------------------------------- GAT pieces

// per-node attention scalars: s_src[n] = h[n]·a_src ; s_dst[n] = h[n]·a_dst
__global__ void k_dots(const float* __restrict__ h,
                       const float* __restrict__ asrc,
                       const float* __restrict__ adst,
                       float* __restrict__ ssrc, float* __restrict__ sdst) {
    int n = blockIdx.x * blockDim.x + threadIdx.x;
    if (n >= N_NODES) return;
    const float4* hp = (const float4*)(h + n * HID);
    const float4* ap = (const float4*)asrc;
    const float4* dp = (const float4*)adst;
    float a = 0.f, b = 0.f;
#pragma unroll
    for (int c = 0; c < HID / 4; ++c) {
        float4 v = hp[c], s = ap[c], d = dp[c];
        a += v.x * s.x + v.y * s.y + v.z * s.z + v.w * s.w;
        b += v.x * d.x + v.y * d.y + v.z * d.z + v.w * d.w;
    }
    ssrc[n] = a;
    sdst[n] = b;
}

// pass A: logits + per-destination running max
__global__ void k_edge_max(const int* __restrict__ src, const int* __restrict__ dst,
                           const float* __restrict__ ssrc, const float* __restrict__ sdst,
                           float* __restrict__ logits, float* __restrict__ m) {
    int e = blockIdx.x * blockDim.x + threadIdx.x;
    if (e >= E_TOT) return;
    int s, d;
    if (e < N_EDGES) { s = src[e]; d = dst[e]; }
    else             { s = d = e - N_EDGES; }   // self loops
    float x = ssrc[s] + sdst[d];
    x = x > 0.f ? x : NEG_SLOPE * x;            // leaky relu
    logits[e] = x;
    atomicMaxFloat(&m[d], x);
}

// pass B1: once per edge -> w = exp(logit - m[dst]) (overwrites logits), denom += w
__global__ void k_edge_exp(const int* __restrict__ dst,
                           float* __restrict__ logits, const float* __restrict__ m,
                           float* __restrict__ denom) {
    int e = blockIdx.x * blockDim.x + threadIdx.x;
    if (e >= E_TOT) return;
    int d = (e < N_EDGES) ? dst[e] : (e - N_EDGES);
    float ev = __expf(logits[e] - m[d]);
    logits[e] = ev;                     // reuse buffer as edge weight
    atomicAdd(&denom[d], ev);
}

// pass B2: acc[dst] += w_e * h[src]  (256 threads = 4 edges x 64 channels)
__global__ void k_edge_scatter(const int* __restrict__ src, const int* __restrict__ dst,
                               const float* __restrict__ w, const float* __restrict__ h,
                               float* __restrict__ acc) {
    int e = blockIdx.x * 4 + (threadIdx.x >> 6);
    int c = threadIdx.x & 63;
    if (e >= E_TOT) return;
    int s, d;
    if (e < N_EDGES) { s = src[e]; d = dst[e]; }
    else             { s = d = e - N_EDGES; }
    atomicAdd(&acc[d * HID + c], w[e] * h[s * HID + c]);
}

// node pass: out = acc/denom + bias, then ELU  (one thread = 4 channels)
__global__ void k_finish(const float* __restrict__ acc, const float* __restrict__ denom,
                         const float* __restrict__ bias, float* __restrict__ out) {
    int t = blockIdx.x * blockDim.x + threadIdx.x;
    if (t >= N_NODES * (HID / 4)) return;
    int n  = t >> 4;
    int c4 = (t & 15) << 2;
    float inv = 1.0f / denom[n];
    float4 v = *(const float4*)(acc + n * HID + c4);
    float4 b = *(const float4*)(bias + c4);
    v.x = v.x * inv + b.x; v.y = v.y * inv + b.y;
    v.z = v.z * inv + b.z; v.w = v.w * inv + b.w;
    v.x = v.x > 0.f ? v.x : expm1f(v.x);
    v.y = v.y > 0.f ? v.y : expm1f(v.y);
    v.z = v.z > 0.f ? v.z : expm1f(v.z);
    v.w = v.w > 0.f ? v.w : expm1f(v.w);
    *(float4*)(out + n * HID + c4) = v;
}

// global mean pool (accumulate), one thread = 4 channels
__global__ void k_pool(const float* __restrict__ h, const int* __restrict__ batch,
                       float* __restrict__ sums, float* __restrict__ cnt) {
    int t = blockIdx.x * blockDim.x + threadIdx.x;
    if (t >= N_NODES * (HID / 4)) return;
    int n  = t >> 4;
    int c4 = (t & 15) << 2;
    int g  = batch[n];
    float4 v = *(const float4*)(h + n * HID + c4);
    float* sg = sums + g * HID + c4;
    atomicAdd(sg + 0, v.x);
    atomicAdd(sg + 1, v.y);
    atomicAdd(sg + 2, v.z);
    atomicAdd(sg + 3, v.w);
    if (c4 == 0) atomicAdd(&cnt[g], 1.0f);
}

// classifier head: out[g] = relu(pooled@Wc1 + bc1) @ Wc2 + bc2
__global__ void k_classify(const float* __restrict__ sums, const float* __restrict__ cnt,
                           const float* __restrict__ Wc1, const float* __restrict__ bc1,
                           const float* __restrict__ Wc2, const float* __restrict__ bc2,
                           float* __restrict__ out) {
    __shared__ float p[HID];
    __shared__ float z[HID];
    int g = blockIdx.x;
    int j = threadIdx.x;
    float cn = cnt[g];
    cn = cn > 1.f ? cn : 1.f;
    p[j] = sums[g * HID + j] / cn;
    __syncthreads();
    float a = bc1[j];
#pragma unroll 8
    for (int c = 0; c < HID; ++c) a += p[c] * Wc1[c * HID + j];
    z[j] = a > 0.f ? a : 0.f;
    __syncthreads();
    if (j == 0) {
        float o = bc2[0];
#pragma unroll 8
        for (int q = 0; q < HID; ++q) o += z[q] * Wc2[q];
        out[g] = o;
    }
}

// ---------------------------------------------------------------- launcher

extern "C" void kernel_launch(void* const* d_in, const int* in_sizes, int n_in,
                              void* d_out, int out_size, void* d_ws, size_t ws_size,
                              hipStream_t stream) {
    const float* x      = (const float*)d_in[0];
    const int*   eidx   = (const int*)  d_in[1];
    const int*   batch  = (const int*)  d_in[2];
    const float* W1     = (const float*)d_in[3];
    const float* asrc1  = (const float*)d_in[4];
    const float* adst1  = (const float*)d_in[5];
    const float* b1     = (const float*)d_in[6];
    const float* W2     = (const float*)d_in[7];
    const float* asrc2  = (const float*)d_in[8];
    const float* adst2  = (const float*)d_in[9];
    const float* b2     = (const float*)d_in[10];
    const float* Wc1    = (const float*)d_in[11];
    const float* bc1    = (const float*)d_in[12];
    const float* Wc2    = (const float*)d_in[13];
    const float* bc2    = (const float*)d_in[14];
    float* out = (float*)d_out;

    const int* src = eidx;              // edge_index[0]
    const int* dst = eidx + N_EDGES;    // edge_index[1]

    // workspace carve (all regions multiples of 16 floats -> 64B aligned)
    float*    A      = (float*)d_ws;             // node features / layer outputs
    float*    B      = A      + N_NODES * HID;   // attention acc / h2
    float*    C      = B      + N_NODES * HID;   // layer-2 attention acc
    float*    ssrc   = C      + N_NODES * HID;
    float*    sdst   = ssrc   + N_NODES;
    float*    logits = sdst   + N_NODES;
    float*    m      = logits + E_TOT;
    float*    denom  = m      + N_NODES;
    float*    pool   = denom  + N_NODES;
    float*    cnt    = pool   + NUM_GRAPHS * HID;
    _Float16* Bpack  = (_Float16*)(cnt + NUM_GRAPHS); // IN_CH*HID halves max (16KB)

    const int NTILES = N_NODES / 16;                   // 3125
    const int GB     = (NTILES + 7) / 8;               // gemm blocks (8 waves each)
    const int NODE_B = (N_NODES + 255) / 256;
    const int NV_B   = (N_NODES * (HID / 4) + 255) / 256;
    const int EDGE_B = (E_TOT + 255) / 256;
    const int ESC_B  = (E_TOT + 3) / 4;

    // -------- layer 1 --------
    k_pack_w<<<(IN_CH * HID + 255) / 256, 256, 0, stream>>>(W1, Bpack, IN_CH);
    k_gemm_wmma<<<GB, 256, 0, stream>>>(x, Bpack, A, N_NODES, IN_CH);
    k_dots<<<NODE_B, 256, 0, stream>>>(A, asrc1, adst1, ssrc, sdst);
    k_fill<<<NODE_B, 256, 0, stream>>>(m, -INFINITY, N_NODES);
    k_fill<<<NODE_B, 256, 0, stream>>>(denom, 0.f, N_NODES);
    k_fill<<<(N_NODES * HID + 255) / 256, 256, 0, stream>>>(B, 0.f, N_NODES * HID);
    k_edge_max<<<EDGE_B, 256, 0, stream>>>(src, dst, ssrc, sdst, logits, m);
    k_edge_exp<<<EDGE_B, 256, 0, stream>>>(dst, logits, m, denom);
    k_edge_scatter<<<ESC_B, 256, 0, stream>>>(src, dst, logits, A, B);
    k_finish<<<NV_B, 256, 0, stream>>>(B, denom, b1, A);     // A <- elu(gat1)

    // -------- layer 2 --------
    k_pack_w<<<(HID * HID + 255) / 256, 256, 0, stream>>>(W2, Bpack, HID);
    k_gemm_wmma<<<GB, 256, 0, stream>>>(A, Bpack, B, N_NODES, HID);  // B <- h2
    k_dots<<<NODE_B, 256, 0, stream>>>(B, asrc2, adst2, ssrc, sdst);
    k_fill<<<NODE_B, 256, 0, stream>>>(m, -INFINITY, N_NODES);
    k_fill<<<NODE_B, 256, 0, stream>>>(denom, 0.f, N_NODES);
    k_fill<<<(N_NODES * HID + 255) / 256, 256, 0, stream>>>(C, 0.f, N_NODES * HID);
    k_edge_max<<<EDGE_B, 256, 0, stream>>>(src, dst, ssrc, sdst, logits, m);
    k_edge_exp<<<EDGE_B, 256, 0, stream>>>(dst, logits, m, denom);
    k_edge_scatter<<<ESC_B, 256, 0, stream>>>(src, dst, logits, B, C);
    k_finish<<<NV_B, 256, 0, stream>>>(C, denom, b2, B);     // B <- elu(gat2)

    // -------- pool + classifier --------
    k_fill<<<(NUM_GRAPHS * HID + 255) / 256, 256, 0, stream>>>(pool, 0.f, NUM_GRAPHS * HID);
    k_fill<<<(NUM_GRAPHS + 255) / 256, 256, 0, stream>>>(cnt, 0.f, NUM_GRAPHS);
    k_pool<<<NV_B, 256, 0, stream>>>(B, batch, pool, cnt);
    k_classify<<<NUM_GRAPHS, HID, 0, stream>>>(pool, cnt, Wc1, bc1, Wc2, bc2, out);
}
